// GCBlock_7799660610109
// MI455X (gfx1250) — matrix operand order repaired and checked by reading, not compiled, not run on hardware
//
#include <hip/hip_runtime.h>
#include <hip/hip_bf16.h>
#include <math.h>

#define D_NODE 256
#define D_MSG  128
#define D_RAD  64
#define HEADS  4
#define OUTC   32

typedef __attribute__((ext_vector_type(16))) __bf16 v16bf;
typedef __attribute__((ext_vector_type(8)))  float  v8f;

__device__ __forceinline__ float bf2f(unsigned short h) {
  return __uint_as_float(((unsigned int)h) << 16);
}
__device__ __forceinline__ unsigned short f2bf_bits(float f) {
  unsigned int u = __float_as_uint(f);
  return (unsigned short)((u + 0x7FFFu + ((u >> 16) & 1u)) >> 16);
}

__device__ __forceinline__ void atomicMaxFloat(float* addr, float val) {
  if (val >= 0.0f) atomicMax((int*)addr, __float_as_int(val));
  else             atomicMin((unsigned int*)addr, __float_as_uint(val));
}

// ---------------------------------------------------------------------------
// D = A @ B^T (+ bias). A:[M,K] f32 row-major. B:[Nout,K] f32 row-major.
// One wave32 computes a 16x32 tile of D (2 column tiles sharing one A
// fragment) with v_wmma_f32_16x16x32_bf16. K compile-time -> fully unrolled.
// M%16==0, Nout%32==0.
// ---------------------------------------------------------------------------
template <int K, bool BF16_OUT>
__global__ void gemm_bf16_wmma(const float* __restrict__ A,
                               const float* __restrict__ B,
                               const float* __restrict__ bias,
                               void* __restrict__ Dst,
                               int M, int Nout) {
  const int lane = threadIdx.x & 31;
  const int wave = (blockIdx.x * blockDim.x + threadIdx.x) >> 5;
  const int colGroups = Nout >> 5;                 // 2 x 16-col tiles per wave
  const int totGroups = (M >> 4) * colGroups;
  if (wave >= totGroups) return;  // wave-uniform: EXEC all-ones for WMMA
  const int rt = wave / colGroups;
  const int cg = wave - rt * colGroups;
  const int row0 = rt << 4;
  const int col0 = cg << 5;
  const int lo  = lane & 15;
  const bool hi = (lane & 16) != 0;

  // A frag (16x32 bf16): lanes 0-15 row M=lo hold K = kk+[0..15];
  // lanes 16-31 same rows, K = kk+16+[0..15]. Contiguous 64B per lane.
  const float* aBase = A + (size_t)(row0 + lo) * K + (hi ? 16 : 0);
  // B frag (32x16 bf16) for column tile t: lanes 0-15 hold W[n, kk+0..15],
  // lanes 16-31 hold W[n, kk+16..31] (contiguous 64B per lane).
  const float* bBase0 = B + (size_t)(col0 + lo) * K + (hi ? 16 : 0);
  const float* bBase1 = bBase0 + (size_t)16 * K;

  v8f acc0 = {};
  v8f acc1 = {};
#pragma unroll
  for (int kk = 0; kk < K; kk += 32) {
    v16bf a, b0, b1;
    const float4 a0 = *(const float4*)(aBase + kk);
    const float4 a1 = *(const float4*)(aBase + kk + 4);
    const float4 a2 = *(const float4*)(aBase + kk + 8);
    const float4 a3 = *(const float4*)(aBase + kk + 12);
    a[0]  = (__bf16)a0.x; a[1]  = (__bf16)a0.y; a[2]  = (__bf16)a0.z; a[3]  = (__bf16)a0.w;
    a[4]  = (__bf16)a1.x; a[5]  = (__bf16)a1.y; a[6]  = (__bf16)a1.z; a[7]  = (__bf16)a1.w;
    a[8]  = (__bf16)a2.x; a[9]  = (__bf16)a2.y; a[10] = (__bf16)a2.z; a[11] = (__bf16)a2.w;
    a[12] = (__bf16)a3.x; a[13] = (__bf16)a3.y; a[14] = (__bf16)a3.z; a[15] = (__bf16)a3.w;
#pragma unroll
    for (int q = 0; q < 4; ++q) {
      const float4 t0 = *(const float4*)(bBase0 + kk + 4 * q);
      const float4 t1 = *(const float4*)(bBase1 + kk + 4 * q);
      b0[4*q+0] = (__bf16)t0.x; b0[4*q+1] = (__bf16)t0.y;
      b0[4*q+2] = (__bf16)t0.z; b0[4*q+3] = (__bf16)t0.w;
      b1[4*q+0] = (__bf16)t1.x; b1[4*q+1] = (__bf16)t1.y;
      b1[4*q+2] = (__bf16)t1.z; b1[4*q+3] = (__bf16)t1.w;
    }
    acc0 = __builtin_amdgcn_wmma_f32_16x16x32_bf16(false, a, false, b0,
                                                   (short)0, acc0, false, false);
    acc1 = __builtin_amdgcn_wmma_f32_16x16x32_bf16(false, a, false, b1,
                                                   (short)0, acc1, false, false);
  }

  // C/D layout: VGPR r -> M = row0 + r (+8 for hi lanes), N = lane&15 of tile
  const int n0 = col0 + lo;
  const int n1 = n0 + 16;
  const float bb0 = bias ? bias[n0] : 0.0f;
  const float bb1 = bias ? bias[n1] : 0.0f;
  const int mBase = row0 + (hi ? 8 : 0);
#pragma unroll
  for (int r = 0; r < 8; ++r) {
    const size_t rowOff = (size_t)(mBase + r) * Nout;
    float v0 = acc0[r] + bb0;
    float v1 = acc1[r] + bb1;
    if (BF16_OUT) {
      ((unsigned short*)Dst)[rowOff + n0] = f2bf_bits(v0);
      ((unsigned short*)Dst)[rowOff + n1] = f2bf_bits(v1);
    } else {
      ((float*)Dst)[rowOff + n0] = v0;
      ((float*)Dst)[rowOff + n1] = v1;
    }
  }
}

// ---------------------------------------------------------------------------
// Wcomb = We @ W_edge  (fuses edge-embed + lin_edge:  e = edge_attr @ Wcomb^T)
// ---------------------------------------------------------------------------
__global__ void combine_w(const float* __restrict__ We,
                          const float* __restrict__ Wedge,
                          float* __restrict__ wcomb) {
  int i = blockIdx.x * blockDim.x + threadIdx.x;
  if (i >= D_MSG * D_RAD) return;
  int o = i >> 6, t = i & 63;
  float s = 0.0f;
#pragma unroll 8
  for (int r = 0; r < D_RAD; ++r) s += We[o * D_RAD + r] * Wedge[r * D_RAD + t];
  wcomb[i] = s;
}

__global__ void init_kernel(float* __restrict__ agg, float* __restrict__ amax,
                            float* __restrict__ asum, int n) {
  int i = blockIdx.x * blockDim.x + threadIdx.x;
  if (i < n * D_MSG) agg[i] = 0.0f;
  if (i < n * HEADS) { amax[i] = -__builtin_inff(); asum[i] = 0.0f; }
}

// ---------------------------------------------------------------------------
// alpha[e,h] = scale * sum_c q[dst,h,c]*(k[src,h,c]+e[e,h,c]);  seg-max(dst)
// ---------------------------------------------------------------------------
__global__ void edge_logits(const int* __restrict__ ei, int E,
                            const float* __restrict__ q,
                            const float* __restrict__ k,
                            const unsigned short* __restrict__ ebf,
                            float* __restrict__ alpha,
                            float* __restrict__ amax) {
  int idx = blockIdx.x * blockDim.x + threadIdx.x;
  if (idx >= E * HEADS) return;
  int e = idx >> 2, h = idx & 3;
  int s = ei[e], d = ei[E + e];
  const float*          qp = q   + (size_t)d * D_MSG + h * OUTC;
  const float*          kp = k   + (size_t)s * D_MSG + h * OUTC;
  const unsigned short* ep = ebf + (size_t)e * D_MSG + h * OUTC;
  float acc = 0.0f;
#pragma unroll 8
  for (int c = 0; c < OUTC; ++c) acc += qp[c] * (kp[c] + bf2f(ep[c]));
  acc *= 0.17677669529663687f;  // 1/sqrt(OUTC)
  alpha[idx] = acc;
  atomicMaxFloat(amax + (size_t)d * HEADS + h, acc);
}

__global__ void edge_exp(const int* __restrict__ ei, int E,
                         float* __restrict__ alpha,
                         const float* __restrict__ amax,
                         float* __restrict__ asum) {
  int idx = blockIdx.x * blockDim.x + threadIdx.x;
  if (idx >= E * HEADS) return;
  int e = idx >> 2, h = idx & 3;
  int d = ei[E + e];
  float ex = expf(alpha[idx] - amax[(size_t)d * HEADS + h]);
  alpha[idx] = ex;
  atomicAdd(asum + (size_t)d * HEADS + h, ex);
}

// ---------------------------------------------------------------------------
// agg[dst] += alpha * (v[src] + e);  L2-resident atomic scatter (51MB << 192MB)
// ---------------------------------------------------------------------------
__global__ void edge_aggregate(const int* __restrict__ ei, int E,
                               const float* __restrict__ v,
                               const unsigned short* __restrict__ ebf,
                               const float* __restrict__ alpha,
                               const float* __restrict__ asum,
                               float* __restrict__ agg) {
  int idx = blockIdx.x * blockDim.x + threadIdx.x;
  if (idx >= E * (D_MSG / 4)) return;
  int e  = idx >> 5;
  int c4 = (idx & 31) << 2;
  int h  = c4 >> 5;
  int s = ei[e], d = ei[E + e];
  float w = alpha[(size_t)e * HEADS + h] /
            (asum[(size_t)d * HEADS + h] + 1e-16f);
  const float*          vp = v   + (size_t)s * D_MSG + c4;
  const unsigned short* ep = ebf + (size_t)e * D_MSG + c4;
  float*                op = agg + (size_t)d * D_MSG + c4;
#pragma unroll
  for (int j = 0; j < 4; ++j)
    atomicAdd(op + j, (vp[j] + bf2f(ep[j])) * w);
}

// ---------------------------------------------------------------------------
// beta = sigmoid(Wbeta . [out, x_r, out-x_r]);  agg <- beta*x_r+(1-beta)*out
// ---------------------------------------------------------------------------
__global__ void beta_gate(const float* __restrict__ Wbeta,
                          const float* __restrict__ xr,
                          float* __restrict__ agg, int n) {
  int i = blockIdx.x * blockDim.x + threadIdx.x;
  if (i >= n) return;
  const float* o = agg + (size_t)i * D_MSG;
  const float* r = xr  + (size_t)i * D_MSG;
  float z = 0.0f;
#pragma unroll 4
  for (int c = 0; c < D_MSG; ++c) {
    float oc = o[c], rc = r[c];
    z += Wbeta[c] * oc + Wbeta[D_MSG + c] * rc + Wbeta[2 * D_MSG + c] * (oc - rc);
  }
  float beta = 1.0f / (1.0f + expf(-z));
  float* ow = agg + (size_t)i * D_MSG;
#pragma unroll 4
  for (int c = 0; c < D_MSG; ++c) {
    float oc = ow[c];
    ow[c] = beta * r[c] + (1.0f - beta) * oc;
  }
}

// ---------------------------------------------------------------------------
extern "C" void kernel_launch(void* const* d_in, const int* in_sizes, int n_in,
                              void* d_out, int out_size, void* d_ws, size_t ws_size,
                              hipStream_t stream) {
  const float* x         = (const float*)d_in[0];
  const float* edge_attr = (const float*)d_in[1];
  const int*   ei        = (const int*)d_in[2];
  const float* W_edge    = (const float*)d_in[3];
  const float* W_n2m     = (const float*)d_in[4];
  const float* Wq = (const float*)d_in[5];   const float* bq = (const float*)d_in[6];
  const float* Wk = (const float*)d_in[7];   const float* bk = (const float*)d_in[8];
  const float* Wv = (const float*)d_in[9];   const float* bv = (const float*)d_in[10];
  const float* We = (const float*)d_in[11];
  const float* Wskip = (const float*)d_in[12]; const float* bskip = (const float*)d_in[13];
  const float* Wbeta = (const float*)d_in[14];
  const float* W_m2n = (const float*)d_in[15];

  const int N = in_sizes[0] / D_NODE;   // 100000
  const int E = in_sizes[1] / D_RAD;    // 1600000

  // workspace layout (floats)
  const size_t nm = (size_t)N * D_MSG;
  float* ws      = (float*)d_ws;
  float* m_buf   = ws;
  float* q_buf   = m_buf  + nm;
  float* k_buf   = q_buf  + nm;
  float* v_buf   = k_buf  + nm;
  float* xr_buf  = v_buf  + nm;
  float* agg     = xr_buf + nm;                     // reused as gated output
  float* amax    = agg    + nm;
  float* asum    = amax   + (size_t)N * HEADS;
  float* wcomb   = asum   + (size_t)N * HEADS;      // [128,64]
  float* alpha   = wcomb  + (size_t)D_MSG * D_RAD;  // [E,4]
  unsigned short* ebf = (unsigned short*)(alpha + (size_t)E * HEADS);  // [E,128] bf16

  auto cdiv = [](long long a, long long b) { return (int)((a + b - 1) / b); };
  const int TB = 256;  // 8 waves/block

  // 0) Wcomb = We @ W_edge (fuses two edge linears)
  combine_w<<<cdiv(D_MSG * D_RAD, TB), TB, 0, stream>>>(We, W_edge, wcomb);

  // 1) init segment buffers
  init_kernel<<<cdiv((long long)N * D_MSG, TB), TB, 0, stream>>>(agg, amax, asum, N);

  // 2) m = x @ W_n2m^T                        [N,256]->[N,128]
  gemm_bf16_wmma<D_NODE, false><<<cdiv((long long)(N / 16) * (D_MSG / 32), 8), TB, 0, stream>>>(
      x, W_n2m, nullptr, m_buf, N, D_MSG);

  // 3) q,k,v,x_r = m @ W^T + b                [N,128]->[N,128] x4
  const int gq = cdiv((long long)(N / 16) * (D_MSG / 32), 8);
  gemm_bf16_wmma<D_MSG, false><<<gq, TB, 0, stream>>>(m_buf, Wq,    bq,    q_buf,  N, D_MSG);
  gemm_bf16_wmma<D_MSG, false><<<gq, TB, 0, stream>>>(m_buf, Wk,    bk,    k_buf,  N, D_MSG);
  gemm_bf16_wmma<D_MSG, false><<<gq, TB, 0, stream>>>(m_buf, Wv,    bv,    v_buf,  N, D_MSG);
  gemm_bf16_wmma<D_MSG, false><<<gq, TB, 0, stream>>>(m_buf, Wskip, bskip, xr_buf, N, D_MSG);

  // 4) e = edge_attr @ Wcomb^T (bf16 out)     [E,64]->[E,128]
  gemm_bf16_wmma<D_RAD, true><<<cdiv((long long)(E / 16) * (D_MSG / 32), 8), TB, 0, stream>>>(
      edge_attr, wcomb, nullptr, ebf, E, D_MSG);

  // 5) attention logits + segment max
  edge_logits<<<cdiv((long long)E * HEADS, TB), TB, 0, stream>>>(ei, E, q_buf, k_buf, ebf, alpha, amax);

  // 6) exp + segment sum
  edge_exp<<<cdiv((long long)E * HEADS, TB), TB, 0, stream>>>(ei, E, alpha, amax, asum);

  // 7) weighted aggregation (atomic scatter into L2-resident agg)
  edge_aggregate<<<cdiv((long long)E * (D_MSG / 4), TB), TB, 0, stream>>>(
      ei, E, v_buf, ebf, alpha, asum, agg);

  // 8) beta-gated skip (in place on agg)
  beta_gate<<<cdiv(N, TB), TB, 0, stream>>>(Wbeta, xr_buf, agg, N);

  // 9) out = gated @ W_m2n^T                  [N,128]->[N,256]
  gemm_bf16_wmma<D_MSG, false><<<cdiv((long long)(N / 16) * (D_NODE / 32), 8), TB, 0, stream>>>(
      agg, W_m2n, nullptr, (float*)d_out, N, D_NODE);
}
